// PreTAttention_40664750358568
// MI455X (gfx1250) — matrix operand order repaired and checked by reading, not compiled, not run on hardware
//
#include <hip/hip_runtime.h>
#include <hip/hip_bf16.h>

typedef __attribute__((ext_vector_type(16))) __bf16 v16bf;
typedef __attribute__((ext_vector_type(8)))  __bf16 v8bf;
typedef __attribute__((ext_vector_type(8)))  float  v8f;
typedef __attribute__((vector_size(16)))     int    v4i;   // async-LDS builtin operand

static constexpr int Bsz = 4, SeqN = 2048, Dm = 768, Hh = 12, HDim = 64, Pp = 64;
static constexpr int NK = Pp + SeqN;      // 2112 keys
static constexpr int TD = 3 * Dm;         // 2304
static constexpr int Mrows = Bsz * SeqN;  // 8192

// ---------------------------------------------------------------------------
// helpers
// ---------------------------------------------------------------------------
__device__ __forceinline__ __bf16 f2bf(float f) {
  unsigned u = __builtin_bit_cast(unsigned, f);
  unsigned r = u + 0x7FFFu + ((u >> 16) & 1u);     // round-to-nearest-even
  unsigned short h = (unsigned short)(r >> 16);
  return __builtin_bit_cast(__bf16, h);
}

__device__ __forceinline__ v8f vzero8() {
  v8f z = {0.f, 0.f, 0.f, 0.f, 0.f, 0.f, 0.f, 0.f};
  return z;
}

// 16-byte global -> LDS copy; async path when the toolchain exposes it.
#if defined(__has_builtin)
#if __has_builtin(__builtin_amdgcn_global_load_async_to_lds_b128)
#define HAVE_ASYNC_LDS 1
#endif
#endif

__device__ __forceinline__ void cp16_g2l(const void* g, void* l) {
#ifdef HAVE_ASYNC_LDS
  __builtin_amdgcn_global_load_async_to_lds_b128((v4i*)g, (v4i*)l, 0, 0);
#else
  *(uint4*)l = *(const uint4*)g;
#endif
}

__device__ __forceinline__ void wait_async0() {
#ifdef HAVE_ASYNC_LDS
#if __has_builtin(__builtin_amdgcn_s_wait_asynccnt)
  __builtin_amdgcn_s_wait_asynccnt(0);
#else
  asm volatile("s_wait_asynccnt 0" ::: "memory");
#endif
#endif
}

// A-fragment (16x32 bf16): lanes 0-15 row=lane, K in {k0..k0+7, k0+16..k0+23},
// k0 = 0 for lanes 0-15, 8 for lanes 16-31 (ISA 7.12.2 16-bit A layout).
__device__ __forceinline__ v16bf load_a_frag(const __bf16* base, int stride, int lane) {
  const int row = lane & 15;
  const int k0  = (lane < 16) ? 0 : 8;
  const __bf16* p = base + row * stride + k0;
  v8bf x0 = *(const v8bf*)(p);
  v8bf x1 = *(const v8bf*)(p + 16);
  v16bf a;
#pragma unroll
  for (int j = 0; j < 8; ++j) { a[j] = x0[j]; a[8 + j] = x1[j]; }
  return a;
}

// B-fragment (32x16 bf16) from column-major staging ("Bl[n][k]"):
// lane col = lane&15, K in {k0..k0+7, k0+16..k0+23}, k0 = 0 (lanes 0-15) / 8.
__device__ __forceinline__ v16bf load_b_frag(const __bf16* base, int stride, int lane) {
  const int col = lane & 15;
  const int k0  = (lane < 16) ? 0 : 8;
  const __bf16* p = base + col * stride + k0;
  v8bf x0 = *(const v8bf*)(p);
  v8bf x1 = *(const v8bf*)(p + 16);
  v16bf b;
#pragma unroll
  for (int j = 0; j < 8; ++j) { b[j] = x0[j]; b[8 + j] = x1[j]; }
  return b;
}

__device__ __forceinline__ v8f wmma_bf16(v16bf a, v16bf b, v8f c) {
  return __builtin_amdgcn_wmma_f32_16x16x32_bf16(false, a, false, b, (short)0, c,
                                                 false, false);
}

// ---------------------------------------------------------------------------
// kernel 0: fp32 -> bf16 elementwise (prompt K/V)
// ---------------------------------------------------------------------------
__global__ __launch_bounds__(256) void cvt_bf16_kernel(const float* __restrict__ in,
                                                       __bf16* __restrict__ out, int n) {
  int i = blockIdx.x * 256 + threadIdx.x;
  if (i < n) out[i] = f2bf(in[i]);
}

// ---------------------------------------------------------------------------
// 128x128 block GEMM, K-step 32, bf16 WMMA, f32 accumulate, LDS double-buffered
// with register-staged global loads (one __syncthreads per K-step).
//   C[M,Nn] = A[M,K] * Bw[K,Nn] (+ bias)
// ---------------------------------------------------------------------------
template <bool A_BF16, bool ADD_BIAS, bool OUT_BF16>
__global__ __launch_bounds__(256) void gemm128_wmma(const void* __restrict__ Av,
                                                    const float* __restrict__ Bw,
                                                    const float* __restrict__ bias,
                                                    void* __restrict__ Cv,
                                                    int M, int Nn, int K) {
  constexpr int AS = 40;              // padded stride (halves)
  __shared__ __bf16 Al[2][128 * AS];  // A tile, row-major [row][k]
  __shared__ __bf16 Bl[2][128 * AS];  // B tile, col-major [n][k]

  const int tid = threadIdx.x, lane = tid & 31, wid = tid >> 5;
  const int wr = wid >> 1, wc = wid & 1;  // 4x2 wave grid
  const int row0 = blockIdx.y * 128, col0 = blockIdx.x * 128;

  v8f acc[2][4];
#pragma unroll
  for (int i = 0; i < 2; ++i)
#pragma unroll
    for (int j = 0; j < 4; ++j) acc[i][j] = vzero8();

  float4 ra[4];   // fp32 A staging regs
  uint2  rha[4];  // bf16 A staging regs
  float4 rb[4];   // B staging regs

  auto load_regs = [&](int k0) {
    if (A_BF16) {
      const __bf16* A = (const __bf16*)Av;
#pragma unroll
      for (int ps = 0; ps < 4; ++ps) {
        const int r = ps * 32 + (tid >> 3), kq = (tid & 7) * 4;
        rha[ps] = *(const uint2*)(A + (size_t)(row0 + r) * K + k0 + kq);
      }
    } else {
      const float* A = (const float*)Av;
#pragma unroll
      for (int ps = 0; ps < 4; ++ps) {
        const int r = ps * 32 + (tid >> 3), kq = (tid & 7) * 4;
        ra[ps] = *(const float4*)(A + (size_t)(row0 + r) * K + k0 + kq);
      }
    }
#pragma unroll
    for (int ps = 0; ps < 4; ++ps) {
      const int kr = ps * 8 + (tid >> 5), cq = (tid & 31) * 4;
      rb[ps] = *(const float4*)(Bw + (size_t)(k0 + kr) * Nn + col0 + cq);
    }
  };

  auto store_lds = [&](__bf16* Ad, __bf16* Bd) {
    if (A_BF16) {
#pragma unroll
      for (int ps = 0; ps < 4; ++ps) {
        const int r = ps * 32 + (tid >> 3), kq = (tid & 7) * 4;
        *(uint2*)&Ad[r * AS + kq] = rha[ps];
      }
    } else {
#pragma unroll
      for (int ps = 0; ps < 4; ++ps) {
        const int r = ps * 32 + (tid >> 3), kq = (tid & 7) * 4;
        __bf16* d = &Ad[r * AS + kq];
        d[0] = f2bf(ra[ps].x); d[1] = f2bf(ra[ps].y);
        d[2] = f2bf(ra[ps].z); d[3] = f2bf(ra[ps].w);
      }
    }
#pragma unroll
    for (int ps = 0; ps < 4; ++ps) {
      const int kr = ps * 8 + (tid >> 5), cq = (tid & 31) * 4;
      Bd[(cq + 0) * AS + kr] = f2bf(rb[ps].x);
      Bd[(cq + 1) * AS + kr] = f2bf(rb[ps].y);
      Bd[(cq + 2) * AS + kr] = f2bf(rb[ps].z);
      Bd[(cq + 3) * AS + kr] = f2bf(rb[ps].w);
    }
  };

  const int kT = K / 32;
  int p = 0;
  load_regs(0);
  store_lds(Al[0], Bl[0]);

  for (int kt = 0; kt < kT; ++kt) {
    __syncthreads();  // buffer p fully staged; previous reads of 1-p done
    const bool more = (kt + 1 < kT);
    if (more) {
      load_regs((kt + 1) * 32);  // global loads in flight behind the WMMAs
      __builtin_prefetch((const float*)Bw + (size_t)((kt + 2) * 32 + (tid >> 5)) * Nn + col0, 0, 0);
    }

    v16bf af[2], bfv[4];
#pragma unroll
    for (int i = 0; i < 2; ++i)
      af[i] = load_a_frag(&Al[p][(wr * 32 + i * 16) * AS], AS, lane);
#pragma unroll
    for (int j = 0; j < 4; ++j)
      bfv[j] = load_b_frag(&Bl[p][(wc * 64 + j * 16) * AS], AS, lane);
#pragma unroll
    for (int i = 0; i < 2; ++i)
#pragma unroll
      for (int j = 0; j < 4; ++j) acc[i][j] = wmma_bf16(af[i], bfv[j], acc[i][j]);

    if (more) store_lds(Al[1 - p], Bl[1 - p]);
    p ^= 1;
  }

  // epilogue: C/D layout = VGPR r -> row r (lanes 0-15) / r+8 (lanes 16-31)
#pragma unroll
  for (int i = 0; i < 2; ++i)
#pragma unroll
    for (int j = 0; j < 4; ++j)
#pragma unroll
      for (int r = 0; r < 8; ++r) {
        const int row = row0 + wr * 32 + i * 16 + ((lane < 16) ? r : r + 8);
        const int col = col0 + wc * 64 + j * 16 + (lane & 15);
        float v = acc[i][j][r];
        if (ADD_BIAS) v += bias[col];
        if (OUT_BF16)
          ((__bf16*)Cv)[(size_t)row * Nn + col] = f2bf(v);
        else
          ((float*)Cv)[(size_t)row * Nn + col] = v;
      }
}

// ---------------------------------------------------------------------------
// Fused flash-attention: block = one (b,h), 128 queries (16 per wave).
// qkv: bf16 [B*N, 3*D] row-major; pro: bf16 [B,2,P,H,HD]; outa: bf16 [B*N, D]
// ---------------------------------------------------------------------------
__global__ __launch_bounds__(256) void attn_fused(const __bf16* __restrict__ qkv,
                                                  const __bf16* __restrict__ pro,
                                                  __bf16* __restrict__ outa) {
  constexpr int QS = 72;  // padded strides (halves)
  __shared__ __bf16 Ql[128 * QS];      // Q tile  [q][hd]
  __shared__ __bf16 Kl[64 * QS];       // K tile  [key][hd]  (== Bl for S)
  __shared__ __bf16 Vl[64 * QS];       // V tile transposed [hd][key] (== Bl for PV)
  __shared__ __bf16 Pl[8 * 16 * QS];   // per-wave P tile [16][64] row-major

  const int tid = threadIdx.x, lane = tid & 31, wid = tid >> 5;
  const int bh = blockIdx.y;
  const int b = bh / Hh, h = bh % Hh;
  const int q0 = blockIdx.x * 128;
  const float scale = 0.125f;  // HD^-0.5

  // --- stage Q once (async global->LDS, 2 threads/row x 64B) ---
  {
    const int r = tid >> 1, part = (tid & 1) * 32;
    const __bf16* src = qkv + (size_t)(b * SeqN + q0 + r) * TD + h * HDim + part;
    __bf16* dst = &Ql[r * QS + part];
#pragma unroll
    for (int j = 0; j < 4; ++j) cp16_g2l(src + j * 8, dst + j * 8);
  }
  wait_async0();
  __syncthreads();

  v16bf aq[2];
  aq[0] = load_a_frag(&Ql[(wid * 16) * QS], QS, lane);
  aq[1] = load_a_frag(&Ql[(wid * 16) * QS + 32], QS, lane);

  float m[8], lsum[8];
  v8f o[4];
#pragma unroll
  for (int r = 0; r < 8; ++r) { m[r] = -3.0e38f; lsum[r] = 0.f; }
#pragma unroll
  for (int j = 0; j < 4; ++j) o[j] = vzero8();

  for (int kt = 0; kt < NK / 64; ++kt) {
    const int k0 = kt * 64;
    // --- stage K (waves 0-3, async) and V-transposed (waves 4-7) ---
    if (tid < 128) {
      const int r = tid >> 1, part = (tid & 1) * 32;
      const int kk = k0 + r;
      const __bf16* src =
          (kk < Pp) ? pro + ((size_t)((b * 2 + 0) * Pp + kk) * Hh + h) * HDim + part
                    : qkv + (size_t)(b * SeqN + kk - Pp) * TD + Dm + h * HDim + part;
      __bf16* dst = &Kl[r * QS + part];
#pragma unroll
      for (int j = 0; j < 4; ++j) cp16_g2l(src + j * 8, dst + j * 8);
    } else {
      const int t = tid - 128;
      const int r = t >> 1, part = (t & 1) * 32;
      const int kk = k0 + r;
      const __bf16* src =
          (kk < Pp) ? pro + ((size_t)((b * 2 + 1) * Pp + kk) * Hh + h) * HDim + part
                    : qkv + (size_t)(b * SeqN + kk - Pp) * TD + 2 * Dm + h * HDim + part;
      union { uint4 u[4]; __bf16 e[32]; } buf;
      const uint4* s = (const uint4*)src;
      buf.u[0] = s[0]; buf.u[1] = s[1]; buf.u[2] = s[2]; buf.u[3] = s[3];
#pragma unroll
      for (int j = 0; j < 32; ++j) Vl[(part + j) * QS + r] = buf.e[j];
    }
    wait_async0();
    __syncthreads();

    // --- S = Q * K^T  (16 queries x 64 keys per wave) ---
    v8f s4[4];
#pragma unroll
    for (int j = 0; j < 4; ++j) {
      v8f s = vzero8();
      s = wmma_bf16(aq[0], load_b_frag(&Kl[(j * 16) * QS], QS, lane), s);
      s = wmma_bf16(aq[1], load_b_frag(&Kl[(j * 16) * QS + 32], QS, lane), s);
      s4[j] = s;
    }

    // --- online softmax, per row-slot r (half-wave owns 16 columns) ---
#pragma unroll
    for (int r = 0; r < 8; ++r) {
      float mx = -3.0e38f;
#pragma unroll
      for (int j = 0; j < 4; ++j) mx = fmaxf(mx, s4[j][r]);
      mx *= scale;
      mx = fmaxf(mx, __shfl_xor(mx, 1, 16));
      mx = fmaxf(mx, __shfl_xor(mx, 2, 16));
      mx = fmaxf(mx, __shfl_xor(mx, 4, 16));
      mx = fmaxf(mx, __shfl_xor(mx, 8, 16));
      const float mn = fmaxf(m[r], mx);
      const float al = __expf(m[r] - mn);
      float ps = 0.f;
#pragma unroll
      for (int j = 0; j < 4; ++j) {
        const float p = __expf(s4[j][r] * scale - mn);
        s4[j][r] = p;
        ps += p;
      }
      ps += __shfl_xor(ps, 1, 16);
      ps += __shfl_xor(ps, 2, 16);
      ps += __shfl_xor(ps, 4, 16);
      ps += __shfl_xor(ps, 8, 16);
      lsum[r] = lsum[r] * al + ps;
      m[r] = mn;
#pragma unroll
      for (int jd = 0; jd < 4; ++jd) o[jd][r] *= al;
      // spill P row to per-wave LDS in A-fragment (row-major) layout
      const int rowL = (lane < 16) ? r : r + 8;
      const int cb = lane & 15;
#pragma unroll
      for (int j = 0; j < 4; ++j)
        Pl[(wid * 16 + rowL) * QS + j * 16 + cb] = f2bf(s4[j][r]);
    }
    // wave-local: make P stores visible to this wave's fragment loads
    asm volatile("s_wait_dscnt 0" ::: "memory");
    __builtin_amdgcn_wave_barrier();

    // --- O += P * V ---
    v16bf ap0 = load_a_frag(&Pl[(wid * 16) * QS], QS, lane);
    v16bf ap1 = load_a_frag(&Pl[(wid * 16) * QS + 32], QS, lane);
#pragma unroll
    for (int jd = 0; jd < 4; ++jd) {
      o[jd] = wmma_bf16(ap0, load_b_frag(&Vl[(jd * 16) * QS], QS, lane), o[jd]);
      o[jd] = wmma_bf16(ap1, load_b_frag(&Vl[(jd * 16) * QS + 32], QS, lane), o[jd]);
    }
    __syncthreads();
  }

  // --- normalize and store bf16 attention output [B*N, D] ---
#pragma unroll
  for (int r = 0; r < 8; ++r) {
    const float inv = 1.0f / lsum[r];
    const int rowL = (lane < 16) ? r : r + 8;
    const size_t row = (size_t)(b * SeqN + q0 + wid * 16 + rowL);
#pragma unroll
    for (int jd = 0; jd < 4; ++jd) {
      const int col = h * HDim + jd * 16 + (lane & 15);
      outa[row * Dm + col] = f2bf(o[jd][r] * inv);
    }
  }
}

// ---------------------------------------------------------------------------
// launch
// ---------------------------------------------------------------------------
extern "C" void kernel_launch(void* const* d_in, const int* in_sizes, int n_in,
                              void* d_out, int out_size, void* d_ws, size_t ws_size,
                              hipStream_t stream) {
  (void)in_sizes; (void)n_in; (void)out_size; (void)ws_size;
  const float* x      = (const float*)d_in[0];
  const float* prompt = (const float*)d_in[1];
  const float* Wqkv   = (const float*)d_in[2];
  const float* Wproj  = (const float*)d_in[3];
  const float* bproj  = (const float*)d_in[4];
  float* out = (float*)d_out;

  __bf16* qkv_bf  = (__bf16*)d_ws;                              // [8192, 2304]
  __bf16* pro_bf  = qkv_bf + (size_t)Mrows * TD;                // [B,2,P,H,HD]
  __bf16* attn_bf = pro_bf + (size_t)Bsz * 2 * Pp * Hh * HDim;  // [8192, 768]

  const int nPrompt = Bsz * 2 * Pp * Hh * HDim;  // 393216
  cvt_bf16_kernel<<<dim3(nPrompt / 256), 256, 0, stream>>>(prompt, pro_bf, nPrompt);

  gemm128_wmma<false, false, true>
      <<<dim3(TD / 128, Mrows / 128), 256, 0, stream>>>(x, Wqkv, nullptr, qkv_bf,
                                                        Mrows, TD, Dm);

  attn_fused<<<dim3(SeqN / 128, Bsz * Hh), 256, 0, stream>>>(qkv_bf, pro_bf, attn_bf);

  gemm128_wmma<true, true, false>
      <<<dim3(Dm / 128, Mrows / 128), 256, 0, stream>>>(attn_bf, Wproj, bproj, out,
                                                        Mrows, Dm, Dm);
}
